// WACBlock_78606491451729
// MI455X (gfx1250) — matrix operand order repaired and checked by reading, not compiled, not run on hardware
//
#include <hip/hip_runtime.h>

typedef __attribute__((ext_vector_type(16))) _Float16 v16h;
typedef __attribute__((ext_vector_type(8)))  _Float16 v8h;
typedef __attribute__((ext_vector_type(8)))  float    v8f;
typedef __attribute__((ext_vector_type(4)))  int      v4i;

#define CCH     192
#define HW      192
#define BATCH   8
#define PLANE   (HW*HW)          /* 36864  */
#define CSTRIDE (CCH*PLANE)      /* 7077888 per-batch */
#define QSCALE  0.14433756729740643f   /* 1/sqrt(48) */

#if __has_builtin(__builtin_amdgcn_global_load_async_to_lds_b128)
#define WAC_ASYNC 1
typedef __attribute__((address_space(1))) v4i* as1_v4i;
typedef __attribute__((address_space(3))) v4i* as3_v4i;
#endif

/* ---------------- workspace layout (bytes) ---------------- */
static constexpr size_t OFF_WQKV = 0;                               // 576*192 f16
static constexpr size_t OFF_WPROJ = OFF_WQKV + (size_t)576*192*2;   // 192*192 f16
static constexpr size_t OFF_W1   = OFF_WPROJ + (size_t)192*192*2;   // 384*192 f16
static constexpr size_t OFF_W2   = OFF_W1   + (size_t)384*192*2;    // 9*192*192 f16 [tap][oc][ic]
static constexpr size_t OFF_BIAS = OFF_W2   + (size_t)9*192*192*2;  // 64*64 f32
static constexpr size_t OFF_G    = OFF_BIAS + (size_t)64*64*4;      // B*C*H*W f16

/* ---------------- WMMA helpers (gfx1250 wave32) ---------------- */
__device__ __forceinline__ v8f wmma16(v16h a, v16h b, v8f c) {
  return __builtin_amdgcn_wmma_f32_16x16x32_f16(false, a, false, b, (short)0, c,
                                                false, false);
}

/* A fragment 16x32, source row-major [m][k], ld multiple of 8 halves. */
__device__ __forceinline__ v16h frag_a(const _Float16* base, int ld, int m0, int k0) {
  const int lane = threadIdx.x & 31;
  const _Float16* p = base + (m0 + (lane & 15)) * ld + k0 + ((lane >> 4) << 3);
  union { v16h v; v8h h[2]; } u;
  u.h[0] = *(const v8h*)(p);
  u.h[1] = *(const v8h*)(p + 16);
  return u.v;
}

/* B fragment 32x16, source stored as [n][k] row-major (weight layout), ld mult 8. */
__device__ __forceinline__ v16h frag_b(const _Float16* base, int ld, int n0, int k0) {
  const int lane = threadIdx.x & 31;
  const _Float16* p = base + (n0 + (lane & 15)) * ld + k0 + ((lane >> 4) << 3);
  union { v16h v; v8h h[2]; } u;
  u.h[0] = *(const v8h*)(p);
  u.h[1] = *(const v8h*)(p + 16);
  return u.v;
}

/* ================= kernel 0: weight conversion + bias gather ================= */
__global__ __launch_bounds__(256) void wac_prep(
    const float* __restrict__ qkv_w, const float* __restrict__ proj_w,
    const float* __restrict__ w1_w,  const float* __restrict__ w2_w,
    const float* __restrict__ bias_table,
    _Float16* __restrict__ wqkv, _Float16* __restrict__ wproj,
    _Float16* __restrict__ w1h,  _Float16* __restrict__ w2h,
    float* __restrict__ biasNN)
{
  const int tid = blockIdx.x * blockDim.x + threadIdx.x;
  const int stride = gridDim.x * blockDim.x;
  for (int i = tid; i < 576*192; i += stride) wqkv[i]  = (_Float16)qkv_w[i];
  for (int i = tid; i < 192*192; i += stride) wproj[i] = (_Float16)proj_w[i];
  for (int i = tid; i < 384*192; i += stride) w1h[i]   = (_Float16)w1_w[i];
  for (int i = tid; i < 9*192*192; i += stride) {
    int tap = i / 36864, r = i % 36864;
    int oc = r / 192, ic = r % 192;
    int ky = tap / 3, kx = tap % 3;
    w2h[i] = (_Float16)w2_w[((oc*192 + ic)*3 + ky)*3 + kx];
  }
  for (int i = tid; i < 64*64; i += stride) {
    int ti = i >> 6, tj = i & 63;                 // token i attends token j
    int idx = ((ti >> 3) - (tj >> 3) + 7) * 15 + ((ti & 7) - (tj & 7) + 7);
    biasNN[i] = bias_table[idx];
  }
}

/* ================= kernel 1: fused window attention =================
   grid = 4608 windows, block = 256 (8 wave32). LDS ~145KB.           */
__global__ __launch_bounds__(256) void wac_attn(
    const float* __restrict__ x,
    const _Float16* __restrict__ wqkv, const float* __restrict__ bqkv,
    const _Float16* __restrict__ wproj, const float* __restrict__ bproj,
    const float* __restrict__ biasNN,
    float* __restrict__ out)
{
  __shared__ _Float16 xa[64][200];     // window activations; reused as attn-out
  __shared__ _Float16 qs[4][64][64];   // per-head Q (pre-scaled), K-pad zeroed
  __shared__ _Float16 ks[4][64][64];
  __shared__ _Float16 vT[4][48][64];   // V transposed: [dcol][token]
  __shared__ float    sbuf[2][64][64]; // scores for 2 heads; P (f16) overlaid

  const int tid  = threadIdx.x;
  const int wave = tid >> 5;
  const int lane = tid & 31;
  const int l15  = lane & 15;
  const int mb8  = (lane >> 4) << 3;
  const int win  = blockIdx.x;
  const int bb   = win / 576;
  const int wrem = win % 576;
  const int py0  = (wrem / 24) * 8, px0 = (wrem % 24) * 8;
  const long gbase = (long)bb * CSTRIDE;

  __builtin_prefetch(wqkv, 0, 2);
  __builtin_prefetch(wproj, 0, 2);

  // stage x window (64 tokens x 192 ch) as f16; zero q/k K-pads (48..63)
  for (int i = tid; i < 192*64; i += 256) {
    int ch = i >> 6, m = i & 63;
    float v = x[gbase + (long)ch * PLANE + (py0 + (m >> 3)) * HW + px0 + (m & 7)];
    xa[m][ch] = (_Float16)v;
  }
  for (int i = tid; i < 4*64*16; i += 256) {
    int h = i >> 10, r = i & 1023, m = r >> 4, j = r & 15;
    qs[h][m][48 + j] = (_Float16)0.f;
    ks[h][m][48 + j] = (_Float16)0.f;
  }
  __syncthreads();

  // ---- QKV GEMM: (64x192) x (192x576), 144 16x16 tiles over 8 waves ----
  for (int t = wave; t < 144; t += 8) {
    const int mt = t & 3, nt = t >> 2;
    v8f acc = {};
    #pragma unroll
    for (int kk = 0; kk < 6; ++kk)
      acc = wmma16(frag_a(&xa[0][0], 200, mt*16, kk*32),
                   frag_b(wqkv, 192, nt*16, kk*32), acc);
    // tile-uniform destination select (16-wide N tile never crosses 192/48)
    const int which = nt / 12;              // 0=q 1=k 2=v
    const int ntc   = nt - which * 12;
    const int h     = ntc / 3;
    const int dc    = (ntc - h * 3) * 16 + l15;
    const float bv  = bqkv[nt * 16 + l15];
    const int m0    = mt * 16 + mb8;
    if (which == 0) {
      #pragma unroll
      for (int r = 0; r < 8; ++r)
        qs[h][m0 + r][dc] = (_Float16)((acc[r] + bv) * QSCALE);
    } else if (which == 1) {
      #pragma unroll
      for (int r = 0; r < 8; ++r)
        ks[h][m0 + r][dc] = (_Float16)(acc[r] + bv);
    } else {
      #pragma unroll
      for (int r = 0; r < 8; ++r)
        vT[h][dc][m0 + r] = (_Float16)(acc[r] + bv);
    }
  }
  __syncthreads();

  // ---- attention: 2 heads per pass (sbuf has 2 slots), 4 waves per head ----
  for (int pass = 0; pass < 2; ++pass) {
    const int sh = wave >> 2;
    const int h  = pass * 2 + sh;
    const int mt = wave & 3;
    // S = Q K^T + bias
    #pragma unroll
    for (int nt = 0; nt < 4; ++nt) {
      v8f acc = {};
      #pragma unroll
      for (int kk = 0; kk < 2; ++kk)
        acc = wmma16(frag_a(&qs[h][0][0], 64, mt*16, kk*32),
                     frag_b(&ks[h][0][0], 64, nt*16, kk*32), acc);
      #pragma unroll
      for (int r = 0; r < 8; ++r) {
        int m = mt*16 + mb8 + r;
        int n = nt*16 + l15;
        sbuf[sh][m][n] = acc[r] + biasNN[m*64 + n];
      }
    }
    __syncthreads();
    // softmax; write normalized P as f16 overlaid on the (thread-owned) row
    if (tid < 128) {
      int hh = tid >> 6, m = tid & 63;
      float* row = &sbuf[hh][m][0];
      float rv[64];
      #pragma unroll
      for (int j = 0; j < 64; ++j) rv[j] = row[j];
      float mx = rv[0];
      #pragma unroll
      for (int j = 1; j < 64; ++j) mx = fmaxf(mx, rv[j]);
      float s = 0.f;
      #pragma unroll
      for (int j = 0; j < 64; ++j) { rv[j] = __expf(rv[j] - mx); s += rv[j]; }
      float inv = 1.f / s;
      _Float16* prow = reinterpret_cast<_Float16*>(row);
      #pragma unroll
      for (int j = 0; j < 64; ++j) prow[j] = (_Float16)(rv[j] * inv);
    }
    __syncthreads();
    // O = P V  (P rows live at stride 128 halves inside sbuf)
    const _Float16* pbase = reinterpret_cast<const _Float16*>(&sbuf[sh][0][0]);
    #pragma unroll
    for (int nt = 0; nt < 3; ++nt) {
      v8f acc = {};
      #pragma unroll
      for (int kk = 0; kk < 2; ++kk)
        acc = wmma16(frag_a(pbase, 128, mt*16, kk*32),
                     frag_b(&vT[h][0][0], 64, nt*16, kk*32), acc);
      #pragma unroll
      for (int r = 0; r < 8; ++r)
        xa[mt*16 + mb8 + r][h*48 + nt*16 + l15] = (_Float16)acc[r];
    }
    __syncthreads();
  }

  // ---- proj + residual -> out = x + attn ----
  for (int t = wave; t < 48; t += 8) {
    const int mt = t & 3, nt = t >> 2;
    v8f acc = {};
    #pragma unroll
    for (int kk = 0; kk < 6; ++kk)
      acc = wmma16(frag_a(&xa[0][0], 200, mt*16, kk*32),
                   frag_b(wproj, 192, nt*16, kk*32), acc);
    const int n = nt*16 + l15;
    const float bp = bproj[n];
    #pragma unroll
    for (int r = 0; r < 8; ++r) {
      int m = mt*16 + mb8 + r;
      long gi = gbase + (long)n * PLANE + (py0 + (m >> 3)) * HW + px0 + (m & 7);
      out[gi] = x[gi] + acc[r] + bp;
    }
  }
}

/* ================= kernel 2: 1x1 conv + GLU -> g (f16) =================
   x2 tile staged via CDNA5 async global->LDS DMA when available.        */
__global__ __launch_bounds__(256) void wac_glu(
    const float* __restrict__ x2, const _Float16* __restrict__ w1h,
    const float* __restrict__ w1b, _Float16* __restrict__ g)
{
  __shared__ _Float16 xa[64][200];
  __shared__ __align__(16) float hb[64][392];   // GEMM out; also f32 staging
  const int tid = threadIdx.x, wave = tid >> 5, lane = tid & 31;
  const int l15 = lane & 15, mb8 = (lane >> 4) << 3;
  const int t = blockIdx.x;
  const int bb = t / 576, pr = t % 576;
  const int y = pr / 3, xo = (pr % 3) * 64;
  const long base = (long)bb * CSTRIDE + y * HW + xo;

  __builtin_prefetch(w1h, 0, 2);
#ifdef WAC_ASYNC
  {
    float* xstage = &hb[0][0];                 // [ch][p] contiguous, 48KB
    for (int j = tid; j < 3072; j += 256) {    // 3072 x 16B async transfers
      int ch = j >> 4, q = (j & 15) << 2;
      const float* gp = x2 + base + (long)ch * PLANE + q;
      float* lp = xstage + ch * 64 + q;
      __builtin_amdgcn_global_load_async_to_lds_b128(
          (as1_v4i)(void*)gp, (as3_v4i)(void*)lp, 0, 0);
    }
#if __has_builtin(__builtin_amdgcn_s_wait_asynccnt)
    __builtin_amdgcn_s_wait_asynccnt(0);
#else
    asm volatile("s_wait_asynccnt 0" ::: "memory");
#endif
    __syncthreads();
    for (int i = tid; i < 192*64; i += 256) {
      int ch = i >> 6, p = i & 63;
      xa[p][ch] = (_Float16)xstage[ch*64 + p];
    }
  }
#else
  for (int i = tid; i < 192*64; i += 256) {
    int ch = i >> 6, p = i & 63;
    xa[p][ch] = (_Float16)x2[base + (long)ch * PLANE + p];
  }
#endif
  __syncthreads();
  for (int tt = wave; tt < 96; tt += 8) {            // 4 x 24 tiles of 16x16
    const int mt = tt & 3, nt = tt >> 2;
    v8f acc = {};
    #pragma unroll
    for (int kk = 0; kk < 6; ++kk)
      acc = wmma16(frag_a(&xa[0][0], 200, mt*16, kk*32),
                   frag_b(w1h, 192, nt*16, kk*32), acc);
    const int n = nt*16 + l15;
    #pragma unroll
    for (int r = 0; r < 8; ++r)
      hb[mt*16 + mb8 + r][n] = acc[r];
  }
  __syncthreads();
  for (int i = tid; i < 192*64; i += 256) {
    int ch = i >> 6, p = i & 63;
    float a  = hb[p][ch] + w1b[ch];
    float bv = hb[p][ch + 192] + w1b[ch + 192];
    g[base + (long)ch * PLANE + p] = (_Float16)(a / (1.f + __expf(-bv)));
  }
}

/* ================= kernel 3: 3x3 conv (rep-pad) + leaky + residual ========= */
__global__ __launch_bounds__(256) void wac_conv3(
    const _Float16* __restrict__ g, const _Float16* __restrict__ w2h,
    const float* __restrict__ w2b, float* __restrict__ out)
{
  __shared__ __align__(16) _Float16 gin[3][66][200]; // [ky][x(-1..64)][ic]
  const int tid = threadIdx.x, wave = tid >> 5, lane = tid & 31;
  const int l15 = lane & 15, mb8 = (lane >> 4) << 3;
  const int t = blockIdx.x;
  const int bb = t / 576, pr = t % 576;
  const int y = pr / 3, xo = (pr % 3) * 64;
  const long gb = (long)bb * CSTRIDE;

  __builtin_prefetch(w2h, 0, 2);
  for (int i = tid; i < 3*192*66; i += 256) {
    int ky = i / (192*66);
    int r2 = i % (192*66);
    int c = r2 / 66, xi = r2 % 66;
    int gy = min(max(y - 1 + ky, 0), HW - 1);
    int gx = min(max(xo - 1 + xi, 0), HW - 1);
    gin[ky][xi][c] = g[gb + (long)c * PLANE + gy * HW + gx];
  }
  __syncthreads();

  // 48 output tiles (4m x 12n), 6 per wave, 9 taps x 6 k-steps each
  v8f accs[6];
  #pragma unroll
  for (int i6 = 0; i6 < 6; ++i6) {
    const int tt = wave + i6 * 8;
    const int mt = tt & 3, nt = tt >> 2;
    v8f acc = {};
    for (int tap = 0; tap < 9; ++tap) {
      const int ky = tap / 3, kx = tap % 3;
      const _Float16* abase = &gin[ky][0][0] + kx * 200;  // x-shift = row shift
      const _Float16* bbase = w2h + (size_t)tap * 36864;
      #pragma unroll
      for (int kk = 0; kk < 6; ++kk)
        acc = wmma16(frag_a(abase, 200, mt*16, kk*32),
                     frag_b(bbase, 192, nt*16, kk*32), acc);
    }
    accs[i6] = acc;
  }
  __syncthreads();                        // all gin reads done -> recycle as obuf

  float* obuf = reinterpret_cast<float*>(&gin[0][0][0]);  // [n][m], 48KB
  #pragma unroll
  for (int i6 = 0; i6 < 6; ++i6) {
    const int tt = wave + i6 * 8;
    const int mt = tt & 3, nt = tt >> 2;
    const int n = nt*16 + l15;
    const float bsum = w2b[n];
    #pragma unroll
    for (int r = 0; r < 8; ++r) {
      float yv = accs[i6][r] + bsum;
      obuf[n*64 + mt*16 + mb8 + r] = yv >= 0.f ? yv : 0.2f * yv;
    }
  }
  __syncthreads();
  // coalesced residual read-modify-write (owning thread only; attn rewrites
  // d_out every call, so this is graph-replay deterministic)
  for (int i = tid; i < 192*64; i += 256) {
    int n = i >> 6, m = i & 63;
    long gi = gb + (long)n * PLANE + y * HW + xo + m;
    out[gi] = out[gi] + obuf[i];
  }
}

/* ================= host launcher ================= */
extern "C" void kernel_launch(void* const* d_in, const int* in_sizes, int n_in,
                              void* d_out, int out_size, void* d_ws, size_t ws_size,
                              hipStream_t stream) {
  (void)in_sizes; (void)n_in; (void)out_size; (void)ws_size;
  const float* x          = (const float*)d_in[0];
  const float* qkv_w      = (const float*)d_in[1];
  const float* qkv_b      = (const float*)d_in[2];
  const float* proj_w     = (const float*)d_in[3];
  const float* proj_b     = (const float*)d_in[4];
  const float* bias_table = (const float*)d_in[5];
  const float* w1_w       = (const float*)d_in[6];
  const float* w1_b       = (const float*)d_in[7];
  const float* w2_w       = (const float*)d_in[8];
  const float* w2_b       = (const float*)d_in[9];
  float* out = (float*)d_out;
  char*  ws  = (char*)d_ws;

  _Float16* wqkv  = (_Float16*)(ws + OFF_WQKV);
  _Float16* wproj = (_Float16*)(ws + OFF_WPROJ);
  _Float16* w1h   = (_Float16*)(ws + OFF_W1);
  _Float16* w2h   = (_Float16*)(ws + OFF_W2);
  float*    biasNN= (float*)   (ws + OFF_BIAS);
  _Float16* gbuf  = (_Float16*)(ws + OFF_G);

  wac_prep <<<512, 256, 0, stream>>>(qkv_w, proj_w, w1_w, w2_w, bias_table,
                                     wqkv, wproj, w1h, w2h, biasNN);
  wac_attn <<<4608, 256, 0, stream>>>(x, wqkv, qkv_b, wproj, proj_b, biasNN, out);
  wac_glu  <<<4608, 256, 0, stream>>>(out, w1h, w1_b, gbuf);
  wac_conv3<<<4608, 256, 0, stream>>>(gbuf, w2h, w2_b, out);
}
